// RoPEAttention_9612136808458
// MI455X (gfx1250) — matrix operand order repaired and checked by reading, not compiled, not run on hardware
//
#include <hip/hip_runtime.h>
#include <hip/hip_bf16.h>

typedef __attribute__((ext_vector_type(16))) _Float16 v16h;
typedef __attribute__((ext_vector_type(8)))  float    v8f;

union Frag { unsigned int u[8]; v16h h; };

static __device__ __forceinline__ v8f wmma_f16(const Frag& a, const Frag& b, v8f c) {
    return __builtin_amdgcn_wmma_f32_16x16x32_f16(
        false, a.h, false, b.h, (short)0, c, false, false);
}

// Shared-aperture flat pointer: low 32 bits are the wave-relative LDS byte
// offset (ISA 10.2, LDS aperture).
static __device__ __forceinline__ unsigned lds_addr(const void* p) {
    return (unsigned)(unsigned long long)p;
}

// Async DMA: 16 bytes per lane, global -> LDS (tracked by ASYNCcnt).
static __device__ __forceinline__ void async_load_b128(unsigned lds, const void* g) {
    asm volatile("global_load_async_to_lds_b128 %0, %1, off"
                 :: "v"(lds), "v"(g) : "memory");
}
static __device__ __forceinline__ void wait_async0() {
    asm volatile("s_wait_asynccnt 0x0" ::: "memory");
}

// ---------------------------------------------------------------------------
// f32 -> f16 conversion
// ---------------------------------------------------------------------------
__global__ void cvt_f32_f16_kernel(const float* __restrict__ in,
                                   _Float16* __restrict__ out, int n) {
    int i = blockIdx.x * blockDim.x + threadIdx.x;
    if (i < n) out[i] = (_Float16)in[i];
}

// ---------------------------------------------------------------------------
// Tiled f16 WMMA GEMM: C[M][Nn] = A[M][K] * B[K][Nn]   (f32 accumulate)
// Block tile 128(M) x 64(N), K-step 32, 256 threads = 8 waves.
// A tile staged with global_load_async_to_lds_b128; B tile transposed in VGPRs.
// ---------------------------------------------------------------------------
template <bool OUT_F16>
__global__ __launch_bounds__(256)
void gemm_f16_kernel(const _Float16* __restrict__ A,
                     const _Float16* __restrict__ Bm,
                     _Float16* __restrict__ Ch,
                     float* __restrict__ Cf,
                     const float* __restrict__ bias,
                     int M, int K, int Nn) {
    __shared__ alignas(16) _Float16 sA[128 * 32];   // [row][k] halves
    __shared__ alignas(16) _Float16 sB[64 * 32];    // transposed: [n][k]

    const int t  = threadIdx.x;
    const int w  = t >> 5;
    const int l  = t & 31;
    const int hb = l >> 4;     // half-wave: 0 or 1
    const int ln = l & 15;
    const int rowBase = blockIdx.x * 128;
    const int colBase = blockIdx.y * 64;
    const unsigned sAo = lds_addr(sA);

    v8f acc[4];
#pragma unroll
    for (int i = 0; i < 4; i++)
#pragma unroll
        for (int j = 0; j < 8; j++) acc[i][j] = 0.0f;

    for (int kc = 0; kc < K; kc += 32) {
        // Stage A tile: 128 rows x 64B = 512 x 16B chunks, async DMA
#pragma unroll
        for (int i = 0; i < 2; i++) {
            int c = t + i * 256;                  // 0..511: row=c>>2, part=c&3
            async_load_b128(sAo + c * 16,
                            A + (size_t)(rowBase + (c >> 2)) * K + kc + (c & 3) * 8);
        }
        // Stage B tile transposed: [n][k], 2048 halves
#pragma unroll
        for (int i = 0; i < 8; i++) {
            int idx = t + i * 256;
            int k = idx >> 6;    // 0..31
            int n = idx & 63;    // 0..63
            sB[n * 32 + k] = Bm[(size_t)(kc + k) * Nn + colBase + n];
        }
        wait_async0();
        __syncthreads();

        // A fragment: lane row = w*16 + ln; K pairs per ISA 16-bit A 16x32 layout
        const unsigned int* sAd = (const unsigned int*)sA;
        Frag a;
#pragma unroll
        for (int i = 0; i < 8; i++) {
            int kp = (i < 4 ? i : 8 + (i - 4)) + 4 * hb;   // = ka/2
            a.u[i] = sAd[(w * 16 + ln) * 16 + kp];
        }
        const unsigned int* sBd = (const unsigned int*)sB;
#pragma unroll
        for (int tc = 0; tc < 4; tc++) {
            Frag b;
#pragma unroll
            for (int i = 0; i < 8; i++) {
                int kp = (i < 4 ? i : 8 + (i - 4)) + 4 * hb;   // = kb/2
                b.u[i] = sBd[(tc * 16 + ln) * 16 + kp];
            }
            acc[tc] = wmma_f16(a, b, acc[tc]);
        }
        __syncthreads();
    }

    // Epilogue: C/D layout — vgpr r -> row r + 8*hb, lane -> col ln
#pragma unroll
    for (int tc = 0; tc < 4; tc++) {
#pragma unroll
        for (int r = 0; r < 8; r++) {
            int row = rowBase + w * 16 + r + 8 * hb;
            int col = colBase + tc * 16 + ln;
            float v = acc[tc][r];
            if (OUT_F16) {
                Ch[(size_t)row * Nn + col] = (_Float16)v;
            } else {
                Cf[(size_t)row * Nn + col] = v + bias[col];
            }
        }
    }
}

// ---------------------------------------------------------------------------
// RoPE + layout reorg.
// QKVh: [B*N][3072] f16 (cols: q|k|v, each reshaped (H=16, hd=64))
// Writes Qh,Kh [b,h,n,64] f16 (Q pre-scaled by 1/sqrt(64)), Vt [b,h,64,n] f16.
// ---------------------------------------------------------------------------
__global__ void rope_kernel(const _Float16* __restrict__ QKVh,
                            const float* __restrict__ cosv,
                            const float* __restrict__ sinv,
                            _Float16* __restrict__ Qh,
                            _Float16* __restrict__ Kh,
                            _Float16* __restrict__ Vt) {
    int idx = blockIdx.x * blockDim.x + threadIdx.x;   // [0, 4096*16*32)
    int d   = idx & 31;
    int h   = (idx >> 5) & 15;
    int row = idx >> 9;          // 0..4095 = b*2048 + n
    int n   = row & 2047;
    int bb  = row >> 11;

    const _Float16* base = QKVh + (size_t)row * 3072;
    float q0 = (float)base[h * 64 + d];
    float q1 = (float)base[h * 64 + d + 32];
    float k0 = (float)base[1024 + h * 64 + d];
    float k1 = (float)base[1024 + h * 64 + d + 32];
    float v0 = (float)base[2048 + h * 64 + d];
    float v1 = (float)base[2048 + h * 64 + d + 32];
    float c0 = cosv[n * 64 + d],      c1 = cosv[n * 64 + d + 32];
    float s0 = sinv[n * 64 + d],      s1 = sinv[n * 64 + d + 32];

    float qr0 = (q0 * c0 - q1 * s0) * 0.125f;
    float qr1 = (q1 * c1 + q0 * s1) * 0.125f;
    float kr0 = k0 * c0 - k1 * s0;
    float kr1 = k1 * c1 + k0 * s1;

    size_t bh = (size_t)bb * 16 + h;
    size_t qi = (bh * 2048 + n) * 64;
    Qh[qi + d]      = (_Float16)qr0;
    Qh[qi + d + 32] = (_Float16)qr1;
    Kh[qi + d]      = (_Float16)kr0;
    Kh[qi + d + 32] = (_Float16)kr1;
    Vt[(bh * 64 + d) * 2048 + n]      = (_Float16)v0;
    Vt[(bh * 64 + d + 32) * 2048 + n] = (_Float16)v1;
}

// ---------------------------------------------------------------------------
// Block-cooperative flash attention.
// Block = 8 waves = one (b, h, 128-query strip); wave w owns 16 query rows.
// Per 64-key iteration: K/V tiles async-DMA'd to LDS once per block,
// 8 wmma for S = Q*K^T, online softmax, P restaged via LDS, 8 wmma for P*V.
// ---------------------------------------------------------------------------
__global__ __launch_bounds__(256)
void attn_kernel(const _Float16* __restrict__ Qh,
                 const _Float16* __restrict__ Kh,
                 const _Float16* __restrict__ Vt,
                 _Float16* __restrict__ AttnOut) {
    __shared__ alignas(16) _Float16 sK[64 * 64];      // [key][d]   8 KB
    __shared__ alignas(16) _Float16 sV[64 * 64];      // [d][key]   8 KB
    __shared__ alignas(16) _Float16 sP[8][16 * 64];   // per-wave P 16 KB

    const int t  = threadIdx.x;
    const int w  = t >> 5;
    const int l  = t & 31;
    const int hb = l >> 4;
    const int ln = l & 15;

    int strip = blockIdx.x & 15;            // 16 strips of 128 query rows
    int h     = (blockIdx.x >> 4) & 15;
    int bb    = blockIdx.x >> 8;
    size_t bh = (size_t)bb * 16 + h;
    int qrow0 = strip * 128 + w * 16;

    const unsigned int* Qd = (const unsigned int*)Qh;
    const _Float16* Kbase = Kh + bh * (size_t)(2048 * 64);
    const _Float16* Vbase = Vt + bh * (size_t)(64 * 2048);
    const unsigned sKo = lds_addr(sK);
    const unsigned sVo = lds_addr(sV);

    // Q fragments: rows qrow0+m, hd split into two 16x32 frags
    Frag aq[2];
    {
        int qrow = qrow0 + ln;
        size_t rowd = (bh * 2048 + qrow) * 32;   // 32 dwords per 64-half row
#pragma unroll
        for (int f = 0; f < 2; f++)
#pragma unroll
            for (int i = 0; i < 8; i++) {
                int kp = (i < 4 ? i : 8 + (i - 4)) + 4 * hb + f * 16;
                aq[f].u[i] = Qd[rowd + kp];
            }
    }

    v8f accO[4];
    float mprev[8], lsum[8];
#pragma unroll
    for (int i = 0; i < 4; i++)
#pragma unroll
        for (int j = 0; j < 8; j++) accO[i][j] = 0.0f;
#pragma unroll
    for (int r = 0; r < 8; r++) { mprev[r] = -1e30f; lsum[r] = 0.0f; }

    for (int kt = 0; kt < 2048; kt += 64) {
        // Async stage K,V tiles: each 64 rows x 128B = 512 x 16B chunks
#pragma unroll
        for (int i = 0; i < 2; i++) {
            int c = t + i * 256;                  // row = c>>3, part = c&7
            async_load_b128(sKo + c * 16,
                            Kbase + (size_t)(kt + (c >> 3)) * 64 + (c & 7) * 8);
            async_load_b128(sVo + c * 16,
                            Vbase + (size_t)(c >> 3) * 2048 + kt + (c & 7) * 8);
        }
        wait_async0();
        __syncthreads();

        // S = Q * K^T : 4 key sub-tiles of 16
        v8f st[4];
#pragma unroll
        for (int s = 0; s < 4; s++)
#pragma unroll
            for (int j = 0; j < 8; j++) st[s][j] = 0.0f;

        const unsigned int* sKd = (const unsigned int*)sK;
#pragma unroll
        for (int s = 0; s < 4; s++) {
            int key = s * 16 + ln;
            Frag bk0, bk1;
#pragma unroll
            for (int i = 0; i < 8; i++) {
                int kp = (i < 4 ? i : 8 + (i - 4)) + 4 * hb;
                bk0.u[i] = sKd[key * 32 + kp];        // hd 0..31
                bk1.u[i] = sKd[key * 32 + kp + 16];   // hd 32..63
            }
            st[s] = wmma_f16(aq[0], bk0, st[s]);
            st[s] = wmma_f16(aq[1], bk1, st[s]);
        }

        // Online softmax: row (r + 8*hb) lives across 16 lanes of a half
        float alpha[8];
#pragma unroll
        for (int r = 0; r < 8; r++) {
            float mx = fmaxf(fmaxf(st[0][r], st[1][r]), fmaxf(st[2][r], st[3][r]));
            mx = fmaxf(mx, __shfl_xor(mx, 1, 16));
            mx = fmaxf(mx, __shfl_xor(mx, 2, 16));
            mx = fmaxf(mx, __shfl_xor(mx, 4, 16));
            mx = fmaxf(mx, __shfl_xor(mx, 8, 16));
            float mnew = fmaxf(mprev[r], mx);
            float a = __expf(mprev[r] - mnew);
            float p0 = __expf(st[0][r] - mnew);
            float p1 = __expf(st[1][r] - mnew);
            float p2 = __expf(st[2][r] - mnew);
            float p3 = __expf(st[3][r] - mnew);
            float rs = (p0 + p1) + (p2 + p3);
            rs += __shfl_xor(rs, 1, 16);
            rs += __shfl_xor(rs, 2, 16);
            rs += __shfl_xor(rs, 4, 16);
            rs += __shfl_xor(rs, 8, 16);
            lsum[r] = lsum[r] * a + rs;
            mprev[r] = mnew;
            alpha[r] = a;
            st[0][r] = p0; st[1][r] = p1; st[2][r] = p2; st[3][r] = p3;
        }
#pragma unroll
        for (int nt = 0; nt < 4; nt++)
#pragma unroll
            for (int r = 0; r < 8; r++) accO[nt][r] *= alpha[r];

        // Restage P (C/D layout) -> A-frag layout via per-wave LDS
#pragma unroll
        for (int r = 0; r < 8; r++) {
            int m = r + 8 * hb;
#pragma unroll
            for (int s = 0; s < 4; s++)
                sP[w][m * 64 + s * 16 + ln] = (_Float16)st[s][r];
        }
        asm volatile("s_wait_dscnt 0x0" ::: "memory");
        Frag pf[2];
        const unsigned int* sPd = (const unsigned int*)sP[w];
#pragma unroll
        for (int f = 0; f < 2; f++)
#pragma unroll
            for (int i = 0; i < 8; i++) {
                int kp = (i < 4 ? i : 8 + (i - 4)) + 4 * hb;
                pf[f].u[i] = sPd[ln * 32 + kp + f * 16];
            }

        // O += P * V
        const unsigned int* sVd = (const unsigned int*)sV;
#pragma unroll
        for (int nt = 0; nt < 4; nt++) {
            Frag bv0, bv1;
            int d = nt * 16 + ln;
#pragma unroll
            for (int i = 0; i < 8; i++) {
                int kp = (i < 4 ? i : 8 + (i - 4)) + 4 * hb;
                bv0.u[i] = sVd[d * 32 + kp];         // keys 0..31
                bv1.u[i] = sVd[d * 32 + kp + 16];    // keys 32..63
            }
            accO[nt] = wmma_f16(pf[0], bv0, accO[nt]);
            accO[nt] = wmma_f16(pf[1], bv1, accO[nt]);
        }
        __syncthreads();   // protect sK/sV before next async stage
    }

    // Normalize and store f16 to AttnOut [row][h*64+d]
#pragma unroll
    for (int nt = 0; nt < 4; nt++)
#pragma unroll
        for (int r = 0; r < 8; r++) {
            int m = r + 8 * hb;
            int qrow = qrow0 + m;
            int col = h * 64 + nt * 16 + ln;
            float v = accO[nt][r] / lsum[r];
            AttnOut[((size_t)bb * 2048 + qrow) * 1024 + col] = (_Float16)v;
        }
}

// ---------------------------------------------------------------------------
// Launch: x,W -> f16 ; QKV GEMM ; RoPE/reorg ; flash attention ; proj GEMM.
// Workspace (halves): Xh 4.19M | Wqh 3.15M | Wph 1.05M | QKVh 12.58M |
//                     Qh 4.19M | Kh 4.19M | Vt 4.19M | AO 4.19M  => ~72 MB
// ---------------------------------------------------------------------------
extern "C" void kernel_launch(void* const* d_in, const int* in_sizes, int n_in,
                              void* d_out, int out_size, void* d_ws, size_t ws_size,
                              hipStream_t stream) {
    const float* x        = (const float*)d_in[0];
    const float* rope_cos = (const float*)d_in[1];
    const float* rope_sin = (const float*)d_in[2];
    const float* W_qkv    = (const float*)d_in[3];
    const float* W_proj   = (const float*)d_in[4];
    const float* b_proj   = (const float*)d_in[5];
    float* out = (float*)d_out;

    _Float16* Xh   = (_Float16*)d_ws;
    _Float16* Wqh  = Xh   + 4194304;    // 4096*1024
    _Float16* Wph  = Wqh  + 3145728;    // 1024*3072
    _Float16* QKVh = Wph  + 1048576;    // 1024*1024
    _Float16* Qh   = QKVh + 12582912;   // 4096*3072
    _Float16* Kh   = Qh   + 4194304;
    _Float16* Vt   = Kh   + 4194304;
    _Float16* AO   = Vt   + 4194304;

    cvt_f32_f16_kernel<<<4194304 / 256, 256, 0, stream>>>(x, Xh, 4194304);
    cvt_f32_f16_kernel<<<3145728 / 256, 256, 0, stream>>>(W_qkv, Wqh, 3145728);
    cvt_f32_f16_kernel<<<1048576 / 256, 256, 0, stream>>>(W_proj, Wph, 1048576);

    // QKV = Xh @ Wqh : M=4096, K=1024, N=3072 -> f16
    gemm_f16_kernel<true><<<dim3(32, 48), 256, 0, stream>>>(
        Xh, Wqh, QKVh, nullptr, nullptr, 4096, 1024, 3072);

    rope_kernel<<<2097152 / 256, 256, 0, stream>>>(QKVh, rope_cos, rope_sin, Qh, Kh, Vt);

    attn_kernel<<<512, 256, 0, stream>>>(Qh, Kh, Vt, AO);

    // out = AO @ Wph + b : M=4096, K=1024, N=1024 -> f32
    gemm_f16_kernel<false><<<dim3(32, 16), 256, 0, stream>>>(
        AO, Wph, nullptr, out, b_proj, 4096, 1024, 1024);
}
